// CausalMHSARoPE_21440476741892
// MI455X (gfx1250) — compile-verified, matrix operand order checked by reading, not run on hardware
//
#include <hip/hip_runtime.h>
#include <stdint.h>

#define D_MODEL   1024
#define NUM_HEADS 16
#define HEAD_SIZE 64
#define SEQ       4096

typedef __bf16 bf16;
typedef __attribute__((ext_vector_type(16))) __bf16       v16bf;
typedef __attribute__((ext_vector_type(8)))  float        v8f;
typedef __attribute__((ext_vector_type(4)))  unsigned int v4u;
typedef __attribute__((ext_vector_type(8)))  int          v8i_;
typedef __attribute__((ext_vector_type(4)))  int          v4i_;

// ---------- bf16 helpers (round-to-nearest-even) ----------
static __device__ __forceinline__ bf16 f2bf(float f) {
    uint32_t u = __float_as_uint(f);
    uint32_t r = (u + 0x7FFFu + ((u >> 16) & 1u)) >> 16;
    uint16_t h = (uint16_t)r;
    return __builtin_bit_cast(bf16, h);
}

// ---------- WMMA fragment loaders (CDNA5 ISA 7.12.2 layouts) ----------
// A-matrix 16x32 bf16 from row-major [16][ld]:
//   lane l: m = l%16, half = l/16; VGPR r holds K = 8*half+2r (r<4) or 16+8*half+2(r-4)
// Per lane this is 2x 8 contiguous bf16 -> vectorizes to paired b128 loads.
static __device__ __forceinline__ v16bf load_a16x32(const bf16* __restrict__ p, int ld) {
    int lane = threadIdx.x & 31;
    int half = lane >> 4;
    const bf16* row = p + (size_t)(lane & 15) * (size_t)ld;
    v16bf a;
#pragma unroll
    for (int r = 0; r < 8; ++r) {
        int kb = (r < 4) ? (8 * half + 2 * r) : (16 + 8 * half + 2 * (r - 4));
        a[2 * r]     = row[kb];
        a[2 * r + 1] = row[kb + 1];
    }
    return a;
}
static __device__ __forceinline__ v8f wmma_bf16(v16bf a, v16bf b, v8f c) {
    return __builtin_amdgcn_wmma_f32_16x16x32_bf16(false, a, false, b, (short)0, c,
                                                   false, false);
}

// ---------- Tensor Data Mover: generic 2D tile global -> LDS (bf16 elems) ----------
// D# per CDNA5 ISA ch.8. 6-arg builtin (clang-23): (v4u, v8i, v4i, v4i, v8i, i32 cpol).
static __device__ __forceinline__ void tdm_load_2d(uint32_t lds_off, const bf16* gsrc,
                                                   uint32_t tile_d0, uint32_t tile_rows,
                                                   uint32_t tensor_d0, uint32_t tensor_rows,
                                                   uint32_t stride0) {
    uint64_t ga = (uint64_t)(uintptr_t)gsrc;
    v4u g0;
    g0[0] = 1u;                                                // count=1 (valid), user mode
    g0[1] = lds_off;                                           // lds_addr (bytes)
    g0[2] = (uint32_t)ga;                                      // global_addr[31:0]
    g0[3] = (uint32_t)((ga >> 32) & 0x01FFFFFFu) | (2u << 30); // addr[56:32] | type=2
    v8i_ g1;
    g1[0] = (int)(1u << 16);                      // data_size=1 -> 2 bytes/elem
    g1[1] = (int)((tensor_d0 & 0xFFFFu) << 16);   // tensor_dim0[15:0]
    g1[2] = (int)((tensor_d0 >> 16) | ((tensor_rows & 0xFFFFu) << 16));
    g1[3] = (int)((tensor_rows >> 16) | (tile_d0 << 16));      // | tile_dim0
    g1[4] = (int)tile_rows;                       // tile_dim1 | tile_dim2=0
    g1[5] = (int)stride0;                         // tensor_dim0_stride[31:0] (elems)
    g1[6] = 0;
    g1[7] = 0;
    v4i_ z4 = {0, 0, 0, 0};
    v8i_ z8 = {0, 0, 0, 0, 0, 0, 0, 0};
    __builtin_amdgcn_tensor_load_to_lds(g0, g1, z4, z4, z8, 0);
    asm volatile("" ::: "memory");
}
static __device__ __forceinline__ uint32_t lds_addr_of(const void* p) {
    return (uint32_t)(uintptr_t)p;                // low 32 bits of generic ptr = LDS addr
}

// ---------- elementwise f32 -> bf16 ----------
__global__ void cast_f32_bf16(const float* __restrict__ in, bf16* __restrict__ out, int n) {
    int i = blockIdx.x * blockDim.x + threadIdx.x;
    if (i < n) out[i] = f2bf(in[i]);
}

// ---------- GEMM: C[MxN] f32 = A[MxK] bf16 * B[NxK]^T bf16 ----------
// One wave computes a 32x64 tile: 2 A-frags x 4 B-frags = 8 WMMAs per K-step.
// No explicit pipelining (avoids spills); inter-wave parallelism hides latency.
__global__ void gemm_bf16_nt(const bf16* __restrict__ A, const bf16* __restrict__ B,
                             float* __restrict__ C, int M, int N, int K) {
    int w = (int)((blockIdx.x * blockDim.x + threadIdx.x) >> 5);
    int tilesN = N >> 6;
    int total  = (M >> 5) * tilesN;
    if (w >= total) return;                 // wave-uniform guard (EXEC stays full)
    int tm = w / tilesN, tn = w % tilesN;
    int row0 = tm << 5, col0 = tn << 6;

    const bf16* A0 = A + (size_t)row0 * K;
    const bf16* A1 = A0 + (size_t)16 * K;
    const bf16* B0 = B + (size_t)(col0 +  0) * K;
    const bf16* B1 = B + (size_t)(col0 + 16) * K;
    const bf16* B2 = B + (size_t)(col0 + 32) * K;
    const bf16* B3 = B + (size_t)(col0 + 48) * K;

    v8f c00 = {}, c01 = {}, c02 = {}, c03 = {};
    v8f c10 = {}, c11 = {}, c12 = {}, c13 = {};

#pragma unroll 1
    for (int k0 = 0; k0 < K; k0 += 32) {
        v16bf a0 = load_a16x32(A0 + k0, K);
        v16bf a1 = load_a16x32(A1 + k0, K);
        v16bf b0 = load_a16x32(B0 + k0, K);
        v16bf b1 = load_a16x32(B1 + k0, K);
        v16bf b2 = load_a16x32(B2 + k0, K);
        v16bf b3 = load_a16x32(B3 + k0, K);
        c00 = wmma_bf16(a0, b0, c00);
        c01 = wmma_bf16(a0, b1, c01);
        c02 = wmma_bf16(a0, b2, c02);
        c03 = wmma_bf16(a0, b3, c03);
        c10 = wmma_bf16(a1, b0, c10);
        c11 = wmma_bf16(a1, b1, c11);
        c12 = wmma_bf16(a1, b2, c12);
        c13 = wmma_bf16(a1, b3, c13);
    }

    int lane = threadIdx.x & 31;
    int n = lane & 15, mb = (lane >> 4) << 3;
#pragma unroll
    for (int r = 0; r < 8; ++r) {
        size_t rowA = (size_t)(row0 + mb + r) * N + col0;
        size_t rowB = (size_t)(row0 + 16 + mb + r) * N + col0;
        C[rowA +  0 + n] = c00[r];
        C[rowA + 16 + n] = c01[r];
        C[rowA + 32 + n] = c02[r];
        C[rowA + 48 + n] = c03[r];
        C[rowB +  0 + n] = c10[r];
        C[rowB + 16 + n] = c11[r];
        C[rowB + 32 + n] = c12[r];
        C[rowB + 48 + n] = c13[r];
    }
}

// ---------- RoPE + head split ----------
// qkv f32 [S][3*D] -> Qh/Kh bf16 [H][S][64] (rotated), VT bf16 [H][64][S] (transposed)
__global__ void rope_split(const float* __restrict__ qkv, const int* __restrict__ pos,
                           bf16* __restrict__ Q, bf16* __restrict__ Kh,
                           bf16* __restrict__ VT) {
    int t = blockIdx.x * blockDim.x + threadIdx.x;
    if (t >= SEQ * NUM_HEADS * 32) return;
    int p = t & 31;
    int h = (t >> 5) & (NUM_HEADS - 1);
    int s = t >> 9;
    float fpos = (float)pos[s];
    float inv  = __powf(10000.0f, -(float)p * (1.0f / 32.0f));
    float sn, c;
    __sincosf(fpos * inv, &sn, &c);
    const float* row = qkv + (size_t)s * (3 * D_MODEL);
    int e = h * HEAD_SIZE + 2 * p;
    float q0 = row[e],               q1 = row[e + 1];
    float k0 = row[D_MODEL + e],     k1 = row[D_MODEL + e + 1];
    float v0 = row[2 * D_MODEL + e], v1 = row[2 * D_MODEL + e + 1];
    size_t o = ((size_t)h * SEQ + s) * HEAD_SIZE + 2 * p;
    Q[o]      = f2bf(q0 * c - q1 * sn);
    Q[o + 1]  = f2bf(q1 * c + q0 * sn);
    Kh[o]     = f2bf(k0 * c - k1 * sn);
    Kh[o + 1] = f2bf(k1 * c + k0 * sn);
    // transposed V: VT[h][dim][s]
    size_t vt = ((size_t)h * HEAD_SIZE + 2 * p) * SEQ + s;
    VT[vt]       = f2bf(v0);
    VT[vt + SEQ] = f2bf(v1);
}

// ---------- Flash attention: ONE WAVE per workgroup, TDM-staged K/V tiles ----------
// blockDim = 32; h/qtile from blockIdx only -> scalar TDM descriptors.
// K tile: [32 keys][64 dims]; V tile: transposed [64 dims][32 keys] so every
// LDS fragment read is 2x 8 contiguous bf16 (ds_load_b128).
__global__ void flash_attn(const bf16* __restrict__ Q, const bf16* __restrict__ Kh,
                           const bf16* __restrict__ VT, bf16* __restrict__ out) {
    __shared__ bf16  Kbuf[2][32 * 64];
    __shared__ bf16  Vbuf[2][64 * 32];
    __shared__ float sS[16 * 32];
    __shared__ bf16  sP[16 * 32];
    __shared__ float sA[16];
    __shared__ float sL[16];

    const int QT = SEQ / 16;
    int h  = (int)blockIdx.x / QT;
    int qt = (int)blockIdx.x % QT;
    int q0 = qt << 4;
    int lane = threadIdx.x & 31;
    int n = lane & 15, mb = (lane >> 4) << 3;

    const bf16* Qb  = Q  + ((size_t)h * SEQ + q0) * HEAD_SIZE;
    const bf16* Kb0 = Kh + (size_t)h * SEQ * HEAD_SIZE;
    const bf16* Vt0 = VT + (size_t)h * HEAD_SIZE * SEQ;

    v8f o0 = {}, o1 = {}, o2 = {}, o3 = {};
    float m_run = -1e30f, l_run = 0.0f;

    v16bf aq0 = load_a16x32(Qb + 0,  HEAD_SIZE);
    v16bf aq1 = load_a16x32(Qb + 32, HEAD_SIZE);

    int jmax = (q0 + 15) >> 5;

    // prologue: TDM-stage K/V block 0 into buffer 0
    tdm_load_2d(lds_addr_of(&Kbuf[0][0]), Kb0, 64, 32, 64, SEQ, 64);
    tdm_load_2d(lds_addr_of(&Vbuf[0][0]), Vt0, 32, 64, SEQ, 64, SEQ);

#pragma unroll 1
    for (int j = 0; j <= jmax; ++j) {
        int k0  = j << 5;
        int buf = j & 1;
        if (j < jmax) {   // issue next block's tiles into other buffer
            int kn = (j + 1) << 5;
            tdm_load_2d(lds_addr_of(&Kbuf[buf ^ 1][0]),
                        Kb0 + (size_t)kn * HEAD_SIZE, 64, 32, 64, SEQ, 64);
            tdm_load_2d(lds_addr_of(&Vbuf[buf ^ 1][0]),
                        Vt0 + kn, 32, 64, SEQ, 64, SEQ);
            __builtin_amdgcn_s_wait_tensorcnt(2);   // current block's 2 loads done
        } else {
            __builtin_amdgcn_s_wait_tensorcnt(0);
        }
        asm volatile("" ::: "memory");

        const bf16* Kt = &Kbuf[buf][0];
        const bf16* Vt = &Vbuf[buf][0];

        // S = Q * K^T from LDS (keys are the N dim -> NxK layout == A layout)
        v8f c0 = {}, c1 = {};
        c0 = wmma_bf16(aq0, load_a16x32(Kt + 0,            HEAD_SIZE), c0);
        c0 = wmma_bf16(aq1, load_a16x32(Kt + 32,           HEAD_SIZE), c0);
        c1 = wmma_bf16(aq0, load_a16x32(Kt + 16 * 64 + 0,  HEAD_SIZE), c1);
        c1 = wmma_bf16(aq1, load_a16x32(Kt + 16 * 64 + 32, HEAD_SIZE), c1);

        // scale + causal mask, stage scores in LDS
#pragma unroll
        for (int r = 0; r < 8; ++r) {
            int m = mb + r, qg = q0 + m;
            sS[m * 32 + n]      = (k0 + n      <= qg) ? c0[r] * 0.125f : -1e30f;
            sS[m * 32 + 16 + n] = (k0 + 16 + n <= qg) ? c1[r] * 0.125f : -1e30f;
        }
        asm volatile("s_wait_dscnt 0" ::: "memory");

        // online softmax: lanes 0-15 each own one row of 32 scores
        if (lane < 16) {
            int row = lane;
            float mx = m_run;
#pragma unroll
            for (int i = 0; i < 32; ++i) mx = fmaxf(mx, sS[row * 32 + i]);
            float alpha = __expf(m_run - mx);
            float sum = 0.0f;
#pragma unroll
            for (int i = 0; i < 32; ++i) {
                float pv = __expf(sS[row * 32 + i] - mx);
                sum += pv;
                sP[row * 32 + i] = f2bf(pv);
            }
            l_run = l_run * alpha + sum;
            m_run = mx;
            sA[row] = alpha;
        }
        asm volatile("s_wait_dscnt 0" ::: "memory");

        // rescale running output by alpha(row)
#pragma unroll
        for (int r = 0; r < 8; ++r) {
            float al = sA[mb + r];
            o0[r] *= al; o1[r] *= al; o2[r] *= al; o3[r] *= al;
        }
        // O += P * V; V tile is transposed [dim][key], so B^T (NxK) == A layout
        v16bf ap = load_a16x32(&sP[0], 32);
        o0 = wmma_bf16(ap, load_a16x32(Vt +  0 * 32, 32), o0);
        o1 = wmma_bf16(ap, load_a16x32(Vt + 16 * 32, 32), o1);
        o2 = wmma_bf16(ap, load_a16x32(Vt + 32 * 32, 32), o2);
        o3 = wmma_bf16(ap, load_a16x32(Vt + 48 * 32, 32), o3);
    }

    if (lane < 16) sL[lane] = 1.0f / l_run;
    asm volatile("s_wait_dscnt 0" ::: "memory");

#pragma unroll
    for (int r = 0; r < 8; ++r) {
        int row = q0 + mb + r;
        float invl = sL[mb + r];
        bf16* orow = out + (size_t)row * D_MODEL + h * HEAD_SIZE;
        orow[n]      = f2bf(o0[r] * invl);
        orow[16 + n] = f2bf(o1[r] * invl);
        orow[32 + n] = f2bf(o2[r] * invl);
        orow[48 + n] = f2bf(o3[r] * invl);
    }
}

extern "C" void kernel_launch(void* const* d_in, const int* in_sizes, int n_in,
                              void* d_out, int out_size, void* d_ws, size_t ws_size,
                              hipStream_t stream) {
    const float* x    = (const float*)d_in[0];
    const int*   tpos = (const int*)d_in[1];
    const float* wqkv = (const float*)d_in[2];
    const float* wo   = (const float*)d_in[3];
    float* out = (float*)d_out;

    char* ws = (char*)d_ws;
    size_t off = 0;
    bf16* xb = (bf16*)(ws + off);    off += (size_t)SEQ * D_MODEL * 2;
    bf16* wqkvb = (bf16*)(ws + off); off += (size_t)3 * D_MODEL * D_MODEL * 2;
    bf16* wob = (bf16*)(ws + off);   off += (size_t)D_MODEL * D_MODEL * 2;
    float* qkv = (float*)(ws + off); off += (size_t)SEQ * 3 * D_MODEL * 4;
    bf16* Qh = (bf16*)(ws + off);    off += (size_t)SEQ * D_MODEL * 2;
    bf16* Kh = (bf16*)(ws + off);    off += (size_t)SEQ * D_MODEL * 2;
    bf16* VT = (bf16*)(ws + off);    off += (size_t)SEQ * D_MODEL * 2;
    bf16* ao = (bf16*)(ws + off);    off += (size_t)SEQ * D_MODEL * 2;

    // 1) casts to bf16
    {
        int n1 = SEQ * D_MODEL;
        cast_f32_bf16<<<(n1 + 255) / 256, 256, 0, stream>>>(x, xb, n1);
        int n2 = 3 * D_MODEL * D_MODEL;
        cast_f32_bf16<<<(n2 + 255) / 256, 256, 0, stream>>>(wqkv, wqkvb, n2);
        int n3 = D_MODEL * D_MODEL;
        cast_f32_bf16<<<(n3 + 255) / 256, 256, 0, stream>>>(wo, wob, n3);
    }
    // 2) QKV projection: [4096,3072] = xb @ Wqkv^T
    {
        int waves = (SEQ / 32) * ((3 * D_MODEL) / 64);
        int blocks = (waves * 32 + 255) / 256;
        gemm_bf16_nt<<<blocks, 256, 0, stream>>>(xb, wqkvb, qkv, SEQ, 3 * D_MODEL, D_MODEL);
    }
    // 3) RoPE + head split (+ transposed V)
    {
        int n = SEQ * NUM_HEADS * 32;
        rope_split<<<(n + 255) / 256, 256, 0, stream>>>(qkv, tpos, Qh, Kh, VT);
    }
    // 4) causal flash attention (1 wave per workgroup, TDM-staged tiles)
    {
        int blocks = NUM_HEADS * (SEQ / 16);
        flash_attn<<<blocks, 32, 0, stream>>>(Qh, Kh, VT, ao);
    }
    // 5) output projection: out = ao @ Wo^T  (f32 result)
    {
        int waves = (SEQ / 32) * (D_MODEL / 64);
        int blocks = (waves * 32 + 255) / 256;
        gemm_bf16_nt<<<blocks, 256, 0, stream>>>(ao, wob, out, SEQ, D_MODEL, D_MODEL);
    }
}
